// BevBackBone_19911468384851
// MI455X (gfx1250) — matrix operand order repaired and checked by reading, not compiled, not run on hardware
//
#include <hip/hip_runtime.h>
#include <hip/hip_bf16.h>

typedef __attribute__((ext_vector_type(16))) _Float16 v16h;
typedef __attribute__((ext_vector_type(8)))  float    v8f;
typedef __attribute__((ext_vector_type(4)))  unsigned int v4u;

union Frag16 { v16h v; v4u q[2]; };

// ---------------------------------------------------------------------------
// Generic implicit-GEMM conv (kh x kh, stride, pad) with WMMA f16 -> f32 acc.
// Double-buffered LDS pipeline: one barrier per 32-K chunk; staging of chunk
// i+1 overlaps the WMMAs consuming chunk i.
// out[co,pos] = post( relu( scale[co]*acc + shift[co] + rpre[co,pos] ) + rpost )
// WG: 256 thr = 8 waves; tile 64(Cout) x 128(pos); each wave 16x64 (4 frags).
// ---------------------------------------------------------------------------
__global__ __launch_bounds__(256)
void k_conv_wmma(const float* __restrict__ Wt, const _Float16* __restrict__ In,
                 int Cin, int H, int W, int Cout, int kh, int stride, int pad,
                 int Ho, int Wo,
                 const float* __restrict__ scale, const float* __restrict__ shift,
                 const _Float16* __restrict__ rpre, const _Float16* __restrict__ rpost,
                 int do_relu, _Float16* __restrict__ obf, float* __restrict__ of32)
{
    __shared__ _Float16 lA[2][64 * 40];   // [co][k], padded row = 40 halves (80B)
    __shared__ _Float16 lB[2][128 * 40];  // [pos][k] (transposed), padded row = 40

    const int tid  = threadIdx.x;
    const int lane = tid & 31, wave = tid >> 5;
    const int wr = wave >> 1, wc = wave & 1;         // 4 x 2 wave grid
    const int n0 = blockIdx.x * 128, m0 = blockIdx.y * 64;
    const int khk = kh * kh;
    const int K   = Cin * khk;
    const int Kt  = (K + 31) & ~31;
    const int N   = Ho * Wo;

    v8f acc[4] = {};

    auto stage = [&](int buf, int k0) {
        // ---- A tile: 64 co x 32 k  (weights f32 -> f16) ----
        for (int i = tid; i < 64 * 32; i += 256) {
            int r = i >> 5, c = i & 31;
            int co = m0 + r, k = k0 + c;
            float w = (co < Cout && k < K) ? Wt[(size_t)co * K + k] : 0.0f;
            lA[buf][r * 40 + c] = (_Float16)w;
        }
        // ---- B tile: 32 k x 128 pos (implicit im2col gather) ----
        for (int i = tid; i < 128 * 32; i += 256) {
            int p = i >> 5, c = i & 31;
            int k = k0 + c, pos = n0 + p;
            float v = 0.0f;
            if (k < K && pos < N) {
                int ci, dy, dx;
                if (kh == 1) { ci = k; dy = 0; dx = 0; }
                else { ci = k / 9; int t = k - ci * 9; dy = t / 3; dx = t - dy * 3; }
                int oy = pos / Wo, ox = pos - oy * Wo;
                int iy = oy * stride + dy - pad;
                int ix = ox * stride + dx - pad;
                if (iy >= 0 && iy < H && ix >= 0 && ix < W)
                    v = (float)In[((size_t)ci * H + iy) * W + ix];
            }
            lB[buf][p * 40 + c] = (_Float16)v;
        }
    };

    const int nk = Kt >> 5;
    stage(0, 0);
    __syncthreads();
    for (int it = 0; it < nk; ++it) {
        const int cur = it & 1;
        if (it + 1 < nk) stage(1 - cur, (it + 1) << 5);   // overlap with WMMAs below

        const int lh = lane >> 4;
        Frag16 fa;
        {
            int row = wr * 16 + (lane & 15);
            fa.q[0] = *(const v4u*)(lA[cur] + row * 40 + 8 * lh);       // K = 8*lh+0..7
            fa.q[1] = *(const v4u*)(lA[cur] + row * 40 + 16 + 8 * lh);  // K = 16+8*lh+0..7
        }
        #pragma unroll
        for (int f = 0; f < 4; ++f) {
            Frag16 fb;
            int prow = wc * 64 + f * 16 + (lane & 15);                  // N = lane%16
            fb.q[0] = *(const v4u*)(lB[cur] + prow * 40 + 8 * lh);      // K = 8*lh+0..7
            fb.q[1] = *(const v4u*)(lB[cur] + prow * 40 + 16 + 8 * lh); // K = 16+8*lh+0..7
            acc[f] = __builtin_amdgcn_wmma_f32_16x16x32_f16(
                false, fa.v, false, fb.v, (short)0, acc[f], false, false);
        }
        __syncthreads();   // single barrier per chunk (ping-pong safe)
    }

    // ---- fused epilogue: bn/bias + residual(pre) + relu + residual(post) ----
    #pragma unroll
    for (int f = 0; f < 4; ++f) {
        int pos = n0 + wc * 64 + f * 16 + (lane & 15);
        if (pos >= N) continue;
        #pragma unroll
        for (int r = 0; r < 8; ++r) {
            int co = m0 + wr * 16 + r + 8 * (lane >> 4);   // C/D layout: M = r + 8*(lane/16)
            if (co >= Cout) continue;
            float v = acc[f][r];
            if (scale) v *= scale[co];
            if (shift) v += shift[co];
            size_t o = (size_t)co * N + pos;
            if (rpre)  v += (float)rpre[o];
            if (do_relu) v = v > 0.0f ? v : 0.0f;
            if (rpost) v += (float)rpost[o];
            if (of32) of32[o] = v;
            else      obf[o]  = (_Float16)v;
        }
    }
}

// ---------------------------------------------------------------------------
// Squeeze fusion: GEMM M=256, N=57344, K=2048 whose B-tile is gathered on the
// fly from camera features y (mean of 2 projected points), fused bias.
// Same double-buffered pipeline.
// ---------------------------------------------------------------------------
__global__ __launch_bounds__(256)
void k_squeeze_wmma(const float* __restrict__ Wt, const float* __restrict__ bias,
                    const float* __restrict__ y, const int* __restrict__ gidx,
                    _Float16* __restrict__ imf)
{
    __shared__ _Float16 lA[2][64 * 40];
    __shared__ _Float16 lB[2][128 * 40];
    const int tid = threadIdx.x, lane = tid & 31, wave = tid >> 5;
    const int wr = wave >> 1, wc = wave & 1;
    const int n0 = blockIdx.x * 128, m0 = blockIdx.y * 64;
    const int K = 2048, N = 57344;
    v8f acc[4] = {};

    auto stage = [&](int buf, int k0) {
        for (int i = tid; i < 64 * 32; i += 256) {
            int r = i >> 5, c = i & 31;
            lA[buf][r * 40 + c] = (_Float16)Wt[(size_t)(m0 + r) * K + k0 + c];
        }
        for (int i = tid; i < 128 * 32; i += 256) {
            int p = i >> 5, c = i & 31;
            int k = k0 + c, pos = n0 + p;
            int h = k >> 7, ch = k & 127;                  // k = h*128 + channel
            int base = (pos * 16 + h) * 2;
            int ia = gidx[base], ib = gidx[base + 1];
            float v = 0.5f * (y[(size_t)ch * 28830 + ia] + y[(size_t)ch * 28830 + ib]);
            lB[buf][p * 40 + c] = (_Float16)v;
        }
    };

    stage(0, 0);
    __syncthreads();
    for (int it = 0; it < 64; ++it) {                      // 2048 / 32
        const int cur = it & 1;
        if (it + 1 < 64) stage(1 - cur, (it + 1) << 5);

        const int lh = lane >> 4;
        Frag16 fa;
        int row = wr * 16 + (lane & 15);
        fa.q[0] = *(const v4u*)(lA[cur] + row * 40 + 8 * lh);
        fa.q[1] = *(const v4u*)(lA[cur] + row * 40 + 16 + 8 * lh);
        #pragma unroll
        for (int f = 0; f < 4; ++f) {
            Frag16 fb;
            int prow = wc * 64 + f * 16 + (lane & 15);
            fb.q[0] = *(const v4u*)(lB[cur] + prow * 40 + 8 * lh);
            fb.q[1] = *(const v4u*)(lB[cur] + prow * 40 + 16 + 8 * lh);
            acc[f] = __builtin_amdgcn_wmma_f32_16x16x32_f16(
                false, fa.v, false, fb.v, (short)0, acc[f], false, false);
        }
        __syncthreads();
    }
    #pragma unroll
    for (int f = 0; f < 4; ++f) {
        int pos = n0 + wc * 64 + f * 16 + (lane & 15);
        #pragma unroll
        for (int r = 0; r < 8; ++r) {
            int co = m0 + wr * 16 + r + 8 * (lane >> 4);
            imf[(size_t)co * N + pos] = (_Float16)(acc[f][r] + bias[co]);
        }
    }
}

// Precompute gather offsets: round(clip(x2y)) -> flat offset into a 93x310 plane.
__global__ void k_make_idx(const float* __restrict__ x2y, int* __restrict__ gidx, int n)
{
    int i = blockIdx.x * blockDim.x + threadIdx.x;
    if (i >= n) return;
    const float* p = x2y + (size_t)i * 4;
    float r0 = fminf(fmaxf(p[0], 0.f), 92.f),  c0 = fminf(fmaxf(p[1], 0.f), 309.f);
    float r1 = fminf(fmaxf(p[2], 0.f), 92.f),  c1 = fminf(fmaxf(p[3], 0.f), 309.f);
    gidx[2 * i]     = (int)rintf(r0) * 310 + (int)rintf(c0);
    gidx[2 * i + 1] = (int)rintf(r1) * 310 + (int)rintf(c1);
}

__global__ void k_f2h(const float* __restrict__ in, _Float16* __restrict__ out, int n)
{
    int i = blockIdx.x * blockDim.x + threadIdx.x;
    if (i < n) out[i] = (_Float16)in[i];
}

// ConvTranspose2d(k=3,s=2,p=1,op=1) + bias + lateral add; tiny -> direct VALU.
__global__ void k_deconv(const float* __restrict__ Wt, const float* __restrict__ bias,
                         const _Float16* __restrict__ In, int Cin, int H, int W,
                         int Co, const _Float16* __restrict__ lat,
                         _Float16* __restrict__ oh, float* __restrict__ of)
{
    int Ho = 2 * H, Wo = 2 * W;
    int i = blockIdx.x * blockDim.x + threadIdx.x;
    int total = Co * Ho * Wo;
    if (i >= total) return;
    int co = i / (Ho * Wo);
    int rem = i - co * Ho * Wo;
    int oy = rem / Wo, ox = rem - oy * Wo;
    float acc = bias[co];
    for (int ky = 0; ky < 3; ++ky) {
        int d = oy + ky - 1;
        if (d < 0 || (d & 1) || (d >> 1) >= H) continue;
        int iy = d >> 1;
        for (int kx = 0; kx < 3; ++kx) {
            int e = ox + kx - 1;
            if (e < 0 || (e & 1) || (e >> 1) >= W) continue;
            int ix = e >> 1;
            const float* wp = Wt + (size_t)co * Cin * 9 + ky * 3 + kx;
            const _Float16* ip = In + (size_t)iy * W + ix;
            for (int ci = 0; ci < Cin; ++ci)
                acc += wp[(size_t)ci * 9] * (float)ip[(size_t)ci * H * W];
        }
    }
    acc += (float)lat[i];
    if (of) of[i] = acc; else oh[i] = (_Float16)acc;
}

// ---------------------------------------------------------------------------
// Host side
// ---------------------------------------------------------------------------
extern "C" void kernel_launch(void* const* d_in, const int* in_sizes, int n_in,
                              void* d_out, int out_size, void* d_ws, size_t ws_size,
                              hipStream_t stream)
{
    (void)in_sizes; (void)n_in; (void)out_size; (void)ws_size;
    auto P = [&](int i) { return (const float*)d_in[i]; };
    // d_in: 0=x[1,33,512,448] 1=y[1,128,93,310] 2=x2y[1,256,224,16,2,2]
    // params (insertion order): 3..8 stem; 9..29 block2; 30..50 block3;
    // 51..71 block4; 72..92 block5; 93..98 lat1..3 (w,b); 99..102 deconv1,2 (w,b);
    // 103..104 squeeze (w,b); 105.. mlp2..5 {w1,b1,w2,b2}.

    char* ws = (char*)d_ws;
    const size_t MB = 1ull << 20;
    _Float16* XB  = (_Float16*)(ws + 0);         // x(f16) -> later mlp hidden
    _Float16* HB  = (_Float16*)(ws + 16 * MB);   // h -> later t1
    _Float16* CB  = (_Float16*)(ws + 32 * MB);   // c1 -> later c3
    _Float16* IMF = (_Float16*)(ws + 48 * MB);   // imf [256,57344]
    int*      IDX = (int*)     (ws + 78 * MB);   // gather offsets -> later t2
    _Float16* EB  = (_Float16*)(ws + 78 * MB);   // t2 (after squeeze done)
    _Float16* FB  = (_Float16*)(ws + 88 * MB);   // idn
    _Float16* GB  = (_Float16*)(ws + 100 * MB);  // mlp out
    _Float16* C2  = (_Float16*)(ws + 112 * MB);  // c2
    _Float16* C4  = (_Float16*)(ws + 124 * MB);  // c4
    _Float16* C5  = (_Float16*)(ws + 126 * MB);  // c5
    _Float16* L5  = (_Float16*)(ws + 127 * MB);
    _Float16* L4  = (_Float16*)(ws + 128 * MB);
    _Float16* Pp4 = (_Float16*)(ws + 129 * MB);
    _Float16* L3  = (_Float16*)(ws + 130 * MB);

    auto conv = [&](const float* Wt, const _Float16* In, int Cin, int H, int W,
                    int Cout, int kh, int stride, int pad,
                    const float* sc, const float* sh,
                    const _Float16* rpre, const _Float16* rpost, int relu,
                    _Float16* obf, float* of32) {
        int Ho = (H + 2 * pad - kh) / stride + 1;
        int Wo = (W + 2 * pad - kh) / stride + 1;
        int N = Ho * Wo;
        dim3 g((N + 127) / 128, (Cout + 63) / 64);
        k_conv_wmma<<<g, dim3(256), 0, stream>>>(Wt, In, Cin, H, W, Cout, kh, stride,
                                                 pad, Ho, Wo, sc, sh, rpre, rpost,
                                                 relu, obf, of32);
    };

    // ---- stem ----
    {
        int n = 33 * 512 * 448;
        k_f2h<<<(n + 255) / 256, 256, 0, stream>>>(P(0), XB, n);
    }
    conv(P(3), XB, 33, 512, 448, 32, 3, 1, 1, P(4), P(5), nullptr, nullptr, 1, HB, nullptr);
    conv(P(6), HB, 32, 512, 448, 32, 3, 1, 1, P(7), P(8), nullptr, nullptr, 1, CB, nullptr);

    // ---- gather indices + fused gather/mean/squeeze GEMM ----
    {
        int n = 57344 * 16;
        k_make_idx<<<(n + 255) / 256, 256, 0, stream>>>(P(2), IDX, n);
        k_squeeze_wmma<<<dim3(448, 4), dim3(256), 0, stream>>>(P(103), P(104),
                                                               P(1), IDX, IMF);
    }

    // ---- four bottleneck stages, each fused with its MLP branch ----
    struct SC { int cin, pl, H, W, sf, pb, pm; };
    const SC sc[4] = {
        { 32, 24, 512, 448, 1,  9, 105},
        { 96, 48, 256, 224, 2, 30, 109},
        {192, 64, 128, 112, 4, 51, 113},
        {256, 96,  64,  56, 8, 72, 117},
    };
    _Float16* ins[4]  = {CB, C2, CB, C4};
    _Float16* outs[4] = {C2, CB, C4, C5};

    for (int s = 0; s < 4; ++s) {
        const SC& c = sc[s];
        int Ho = c.H / 2, Wo = c.W / 2, co4 = 4 * c.pl;
        int pb = c.pb, p1 = c.pb + 12, pm = c.pm;
        _Float16* in = ins[s];
        _Float16* out = outs[s];
        // MLP branch: imf[::sf,::sf] -> 1x1(256->128) -> 1x1(128->co4)
        conv(P(pm + 0), IMF, 256, 256, 224, 128, 1, c.sf, 0,
             nullptr, P(pm + 1), nullptr, nullptr, 0, XB, nullptr);
        conv(P(pm + 2), XB, 128, Ho, Wo, co4, 1, 1, 0,
             nullptr, P(pm + 3), nullptr, nullptr, 0, GB, nullptr);
        // bottleneck 0 (stride 2, with downsample)
        conv(P(pb + 0), in, c.cin, c.H, c.W, c.pl, 1, 1, 0,
             P(pb + 1), P(pb + 2), nullptr, nullptr, 1, HB, nullptr);
        conv(P(pb + 3), HB, c.pl, c.H, c.W, c.pl, 3, 2, 1,
             P(pb + 4), P(pb + 5), nullptr, nullptr, 1, EB, nullptr);
        conv(P(pb + 9), in, c.cin, c.H, c.W, co4, 1, 2, 0,
             P(pb + 10), P(pb + 11), nullptr, nullptr, 0, FB, nullptr);
        conv(P(pb + 6), EB, c.pl, Ho, Wo, co4, 1, 1, 0,
             P(pb + 7), P(pb + 8), FB, nullptr, 1, out, nullptr);
        // bottleneck 1 (stride 1, identity residual) + post-add of MLP branch
        conv(P(p1 + 0), out, co4, Ho, Wo, c.pl, 1, 1, 0,
             P(p1 + 1), P(p1 + 2), nullptr, nullptr, 1, HB, nullptr);
        conv(P(p1 + 3), HB, c.pl, Ho, Wo, c.pl, 3, 1, 1,
             P(p1 + 4), P(p1 + 5), nullptr, nullptr, 1, EB, nullptr);
        conv(P(p1 + 6), EB, c.pl, Ho, Wo, co4, 1, 1, 0,
             P(p1 + 7), P(p1 + 8), out, GB, 1, out, nullptr);
    }

    // ---- FPN head ----
    conv(P(93), C5, 384, 32, 28, 196, 1, 1, 0, nullptr, P(94), nullptr, nullptr, 0, L5, nullptr);
    conv(P(95), C4, 256, 64, 56, 128, 1, 1, 0, nullptr, P(96), nullptr, nullptr, 0, L4, nullptr);
    {
        int total = 128 * 64 * 56;
        k_deconv<<<(total + 255) / 256, 256, 0, stream>>>(P(99), P(100), L5, 196,
                                                          32, 28, 128, L4, Pp4, nullptr);
    }
    conv(P(97), CB, 192, 128, 112, 96, 1, 1, 0, nullptr, P(98), nullptr, nullptr, 0, L3, nullptr);
    {
        int total = 96 * 128 * 112;
        k_deconv<<<(total + 255) / 256, 256, 0, stream>>>(P(101), P(102), Pp4, 128,
                                                          64, 56, 96, L3, nullptr,
                                                          (float*)d_out);
    }
}
